// MambaLayer_71305047048860
// MI455X (gfx1250) — matrix operand order repaired and checked by reading, not compiled, not run on hardware
//
#include <hip/hip_runtime.h>
#include <hip/hip_bf16.h>

// ---------------------------------------------------------------------------
// Problem constants (from reference): B=2, DIM=256, L=16^3=4096, D_INNER=512,
// D_STATE=16, DT_RANK=16, D_CONV=4, DEPTH=2.  TOK = B*L = 8192.
// ---------------------------------------------------------------------------
#define TOK      8192
#define NL       4096
#define DIMC     256
#define DI       512
#define DSTATE   16
#define DTRANK   16
#define XDBL_W   48
#define CONV_K   6912   // 256 * 27

typedef __attribute__((ext_vector_type(16))) __bf16 v16bf;
typedef __attribute__((ext_vector_type(8)))  float  v8f;

__device__ __forceinline__ unsigned short f2bf(float f) {
    union { float f; unsigned u; } cv; cv.f = f;
    unsigned u = cv.u;
    unsigned r = u + 0x7FFFu + ((u >> 16) & 1u);   // round-to-nearest-even
    return (unsigned short)(r >> 16);
}

__device__ __forceinline__ float softplusf(float v) {
    return v > 20.f ? v : log1pf(expf(v));
}

__device__ __forceinline__ float siluf(float v) {
    return v / (1.f + expf(-v));
}

// ---------------------------------------------------------------------------
// fp32 -> bf16 convert
// ---------------------------------------------------------------------------
__global__ __launch_bounds__(256) void convert_bf16_kernel(
    const float* __restrict__ src, unsigned short* __restrict__ dst, int n) {
    int i = blockIdx.x * blockDim.x + threadIdx.x;
    if (i < n) dst[i] = f2bf(src[i]);
}

// ---------------------------------------------------------------------------
// Repack conv3d weights (O,I,27) -> bf16 [co][kidx*256 + ci] so that WMMA B
// fragments are contiguous packed pairs along ci.
// ---------------------------------------------------------------------------
__global__ __launch_bounds__(256) void repack_conv_w_kernel(
    const float* __restrict__ w, unsigned short* __restrict__ wr) {
    int i = blockIdx.x * blockDim.x + threadIdx.x;
    if (i >= DIMC * CONV_K) return;
    int co   = i / CONV_K;
    int rem  = i - co * CONV_K;
    int ci   = rem / 27;
    int kidx = rem - ci * 27;
    wr[(size_t)co * CONV_K + kidx * DIMC + ci] = f2bf(w[i]);
}

// ---------------------------------------------------------------------------
// LayerNorm over channels: x (B,256,L) -> h[token][c], bf16 copy hb.
// One token per block (256 threads).
// ---------------------------------------------------------------------------
__global__ __launch_bounds__(256) void ln_kernel(
    const float* __restrict__ x, const float* __restrict__ w,
    const float* __restrict__ b, float* __restrict__ h,
    unsigned short* __restrict__ hb) {
    int t  = blockIdx.x;              // token
    int bi = t >> 12;                 // batch
    int l  = t & (NL - 1);
    int c  = threadIdx.x;
    float v = x[((size_t)(bi * DIMC + c)) * NL + l];
    __shared__ float red[256];
    red[c] = v; __syncthreads();
    for (int s = 128; s > 0; s >>= 1) { if (c < s) red[c] += red[c + s]; __syncthreads(); }
    float mu = red[0] * (1.f / 256.f);
    __syncthreads();
    float d = v - mu;
    red[c] = d * d; __syncthreads();
    for (int s = 128; s > 0; s >>= 1) { if (c < s) red[c] += red[c + s]; __syncthreads(); }
    float var = red[0] * (1.f / 256.f);
    float o = d * rsqrtf(var + 1e-5f) * w[c] + b[c];
    h[(size_t)t * DIMC + c]  = o;
    hb[(size_t)t * DIMC + c] = f2bf(o);
}

// ---------------------------------------------------------------------------
// Generic WMMA GEMM: C[M][N] (fp32) = A[M][K] (bf16, row-major)
//                                   * W[N][K]^T (bf16, row-major weights).
// One 16x16 C tile per wave. K % 32 == 0, M,N % 16 == 0.
// Fragment layouts per CDNA5 ISA 7.12.2 (16-bit A 16x32, B 32x16, f32 C/D).
// ---------------------------------------------------------------------------
__global__ __launch_bounds__(256) void wmma_gemm_kernel(
    const unsigned short* __restrict__ A, const unsigned short* __restrict__ W,
    float* __restrict__ C, int M, int N, int K) {
    int wave = blockIdx.x * (blockDim.x >> 5) + (threadIdx.x >> 5);
    int lane = threadIdx.x & 31;
    int tilesN = N >> 4;
    int tiles  = (M >> 4) * tilesN;
    if (wave >= tiles) return;
    int tm = wave / tilesN;
    int tn = wave - tm * tilesN;
    int half = lane >> 4;
    int q    = lane & 15;

    const unsigned short* Ap = A + (size_t)(tm * 16 + q) * K + 8 * half;
    const unsigned short* Wp = W + (size_t)(tn * 16 + q) * K + 8 * half;

    v8f acc = {};
    for (int k0 = 0; k0 < K; k0 += 32) {
        if (k0 + 32 < K) {  // gfx1250 global_prefetch_b8 for next K slab
            __builtin_prefetch(Ap + k0 + 32, 0, 1);
            __builtin_prefetch(Wp + k0 + 32, 0, 1);
        }
        union { unsigned u[8]; v16bf v; } fa, fb;
#pragma unroll
        for (int vv = 0; vv < 8; vv++) {
            // A: VGPR v<4 -> K = 2v (+8*half), v>=4 -> K = 16+2(v-4) (+8*half)
            int ka = k0 + ((vv & 3) * 2) + ((vv >> 2) * 16);
            fa.u[vv] = *(const unsigned*)(Ap + ka);
            // B: VGPR v<4 -> K = 2v (+8*half), v>=4 -> K = 16+2(v-4) (+8*half)
            fb.u[vv] = *(const unsigned*)(Wp + ka);
        }
        acc = __builtin_amdgcn_wmma_f32_16x16x32_bf16(
            false, fa.v, false, fb.v, (short)0, acc, false, false);
    }

    float* Cp = C + (size_t)(tm * 16 + 8 * half) * N + tn * 16 + q;
#pragma unroll
    for (int r = 0; r < 8; r++) Cp[(size_t)r * N] = acc[r];
}

// ---------------------------------------------------------------------------
// Depthwise causal conv1d (width 4, left pad 3) + bias + SiLU on u = xz[:,:512]
// ---------------------------------------------------------------------------
__global__ __launch_bounds__(256) void conv_silu_kernel(
    const float* __restrict__ xz, const float* __restrict__ cw,
    const float* __restrict__ cb, float* __restrict__ u,
    unsigned short* __restrict__ ub) {
    int idx = blockIdx.x * blockDim.x + threadIdx.x;   // TOK*DI
    if (idx >= TOK * DI) return;
    int d  = idx & (DI - 1);
    int t  = idx >> 9;
    int bi = t >> 12;
    int l  = t & (NL - 1);
    float acc = cb[d];
#pragma unroll
    for (int j = 0; j < 4; j++) {
        int ls = l - 3 + j;
        if (ls >= 0)
            acc += cw[d * 4 + j] * xz[((size_t)((bi << 12) + ls)) * 1024 + d];
    }
    float s = siluf(acc);
    u[(size_t)t * DI + d]  = s;
    ub[(size_t)t * DI + d] = f2bf(s);
}

// ---------------------------------------------------------------------------
// dt = softplus(dt_in @ dtw^T + dtb);  dt_in = x_dbl[:, :16], dtw (512,16)
// ---------------------------------------------------------------------------
__global__ __launch_bounds__(256) void dt_kernel(
    const float* __restrict__ xdbl, const float* __restrict__ dtw,
    const float* __restrict__ dtb, float* __restrict__ dt) {
    int idx = blockIdx.x * blockDim.x + threadIdx.x;
    if (idx >= TOK * DI) return;
    int d = idx & (DI - 1);
    int t = idx >> 9;
    const float* xi = xdbl + (size_t)t * XDBL_W;
    float acc = dtb[d];
#pragma unroll
    for (int r = 0; r < DTRANK; r++) acc += xi[r] * dtw[d * DTRANK + r];
    dt[(size_t)t * DI + d] = softplusf(acc);
}

// ---------------------------------------------------------------------------
// Selective scan: one thread per (b, d); 16 fp32 states in registers,
// sequential over L = 4096.  y = (scan + u*D) * silu(z).
// ---------------------------------------------------------------------------
__global__ __launch_bounds__(256) void scan_kernel(
    const float* __restrict__ dt, const float* __restrict__ u,
    const float* __restrict__ xdbl, const float* __restrict__ xz,
    const float* __restrict__ A_log, const float* __restrict__ Dskip,
    float* __restrict__ y, unsigned short* __restrict__ yb) {
    int idx = blockIdx.x * blockDim.x + threadIdx.x;
    if (idx >= 2 * DI) return;
    int bi = idx >> 9;
    int d  = idx & (DI - 1);
    float Aneg[DSTATE];
#pragma unroll
    for (int s = 0; s < DSTATE; s++) Aneg[s] = -expf(A_log[d * DSTATE + s]);
    float Dv = Dskip[d];
    float st[DSTATE];
#pragma unroll
    for (int s = 0; s < DSTATE; s++) st[s] = 0.f;

    for (int l = 0; l < NL; l++) {
        int t = (bi << 12) + l;
        float dtv = dt[(size_t)t * DI + d];
        float uv  = u[(size_t)t * DI + d];
        const float* xd = xdbl + (size_t)t * XDBL_W;
        float yv = 0.f;
#pragma unroll
        for (int s = 0; s < DSTATE; s++) {
            float dA = __expf(dtv * Aneg[s]);
            float Bm = xd[DTRANK + s];
            float Cm = xd[DTRANK + DSTATE + s];
            st[s] = dA * st[s] + dtv * Bm * uv;
            yv += st[s] * Cm;
        }
        yv += uv * Dv;
        float zv = xz[(size_t)t * 1024 + DI + d];
        yv *= siluf(zv);
        y[(size_t)t * DI + d]  = yv;
        yb[(size_t)t * DI + d] = f2bf(yv);
    }
}

// ---------------------------------------------------------------------------
// Fused implicit-GEMM conv3d (3x3x3, pad 1) via WMMA + PReLU + softplus(scale)
// + add Mamba output; writes final NCDHW output.
// M = TOK (tokens), N = 256 (out channels), K = 27*256 ordered (kidx, ci) so
// each 32-wide K step has a fixed spatial tap.
// ---------------------------------------------------------------------------
__global__ __launch_bounds__(256) void conv3d_wmma_kernel(
    const unsigned short* __restrict__ xb, const unsigned short* __restrict__ wr,
    const float* __restrict__ hmamba, const float* __restrict__ prelu_a,
    const float* __restrict__ scale, float* __restrict__ out) {
    int wave = blockIdx.x * (blockDim.x >> 5) + (threadIdx.x >> 5);
    int lane = threadIdx.x & 31;
    int tiles = (TOK >> 4) * (DIMC >> 4);   // 512 * 16
    if (wave >= tiles) return;
    int tm = wave >> 4;
    int tn = wave & 15;
    int half = lane >> 4;
    int q    = lane & 15;

    // Token for this lane's A rows
    int tA = tm * 16 + q;
    int bA = tA >> 12;
    int lA = tA & (NL - 1);
    int zA = lA >> 8, yA = (lA >> 4) & 15, xA = lA & 15;

    const unsigned short* Wp = wr + (size_t)(tn * 16 + q) * CONV_K + 8 * half;

    v8f acc = {};
    for (int k0 = 0; k0 < CONV_K; k0 += 32) {
        int kidx = k0 >> 8;            // spatial tap (fixed across the 32-slab)
        int ci0  = k0 & 255;
        int kd = kidx / 9;
        int rem = kidx - kd * 9;
        int kh = rem / 3;
        int kw = rem - kh * 3;
        int zz = zA + kd - 1, yn = yA + kh - 1, xn = xA + kw - 1;
        bool valid = ((unsigned)zz < 16u) & ((unsigned)yn < 16u) & ((unsigned)xn < 16u);
        int ls = (zz << 8) + (yn << 4) + xn;

        if (k0 + 32 < CONV_K) __builtin_prefetch(Wp + k0 + 32, 0, 1);

        union { unsigned u[8]; v16bf v; } fa, fb;
#pragma unroll
        for (int vv = 0; vv < 8; vv++) {
            int cof = ((vv & 3) * 2) + ((vv >> 2) * 16) + 8 * half;
            int ci = ci0 + cof;
            unsigned pa = 0u;
            if (valid) {
                unsigned short lo = xb[(size_t)((bA << 8) + ci)     * NL + ls];
                unsigned short hi = xb[(size_t)((bA << 8) + ci + 1) * NL + ls];
                pa = (unsigned)lo | ((unsigned)hi << 16);
            }
            fa.u[vv] = pa;
            fb.u[vv] = *(const unsigned*)(Wp + k0 + ((vv & 3) * 2) + ((vv >> 2) * 16));
        }
        acc = __builtin_amdgcn_wmma_f32_16x16x32_bf16(
            false, fa.v, false, fb.v, (short)0, acc, false, false);
    }

    float pav = prelu_a[0];
    float sc  = scale[0];
    float sp  = softplusf(sc);
    int co = tn * 16 + q;
#pragma unroll
    for (int r = 0; r < 8; r++) {
        int m   = r + 8 * half;
        int tok = tm * 16 + m;
        int b   = tok >> 12;
        int l   = tok & (NL - 1);
        float cv = acc[r];
        cv = cv > 0.f ? cv : pav * cv;
        out[(size_t)((b << 8) + co) * NL + l] =
            hmamba[(size_t)tok * DIMC + co] + sp * cv;
    }
}

// ---------------------------------------------------------------------------
// Host launcher
// ---------------------------------------------------------------------------
extern "C" void kernel_launch(void* const* d_in, const int* in_sizes, int n_in,
                              void* d_out, int out_size, void* d_ws, size_t ws_size,
                              hipStream_t stream) {
    (void)in_sizes; (void)n_in; (void)out_size; (void)ws_size;
    const float* x         = (const float*)d_in[0];
    const float* ln_w      = (const float*)d_in[1];
    const float* ln_b      = (const float*)d_in[2];
    const float* in_proj_w = (const float*)d_in[3];
    const float* conv_w    = (const float*)d_in[4];
    const float* conv_b    = (const float*)d_in[5];
    const float* x_proj_w  = (const float*)d_in[6];
    const float* dt_proj_w = (const float*)d_in[7];
    const float* dt_proj_b = (const float*)d_in[8];
    const float* A_log     = (const float*)d_in[9];
    const float* D_skip    = (const float*)d_in[10];
    const float* out_proj_w= (const float*)d_in[11];
    const float* conv3d_w  = (const float*)d_in[12];
    const float* prelu_a   = (const float*)d_in[13];
    const float* scale     = (const float*)d_in[14];
    float* out = (float*)d_out;

    char* ws = (char*)d_ws;
    size_t off = 0;
    auto take = [&](size_t bytes) -> char* {
        char* p = ws + off;
        off += (bytes + 255) & ~(size_t)255;
        return p;
    };
    float*          h    = (float*)take((size_t)TOK * DIMC * 4);
    unsigned short* hb   = (unsigned short*)take((size_t)TOK * DIMC * 2);
    float*          xz   = (float*)take((size_t)TOK * 1024 * 4);
    float*          u    = (float*)take((size_t)TOK * DI * 4);
    unsigned short* ub   = (unsigned short*)take((size_t)TOK * DI * 2);
    float*          xdbl = (float*)take((size_t)TOK * XDBL_W * 4);
    float*          dtv  = (float*)take((size_t)TOK * DI * 4);
    float*          y    = (float*)take((size_t)TOK * DI * 4);
    unsigned short* yb   = (unsigned short*)take((size_t)TOK * DI * 2);
    unsigned short* xb   = (unsigned short*)take((size_t)2 * DIMC * NL * 2);
    unsigned short* wr   = (unsigned short*)take((size_t)DIMC * CONV_K * 2);
    unsigned short* wbuf = (unsigned short*)take((size_t)1024 * DIMC * 2);

    const int NX = 2 * DIMC * NL;  // 2,097,152

    // bf16 copy of x for the conv3d implicit GEMM + repacked weights
    convert_bf16_kernel<<<(NX + 255) / 256, 256, 0, stream>>>(x, xb, NX);
    repack_conv_w_kernel<<<(DIMC * CONV_K + 255) / 256, 256, 0, stream>>>(conv3d_w, wr);

    // LayerNorm
    ln_kernel<<<TOK, 256, 0, stream>>>(x, ln_w, ln_b, h, hb);

    for (int dp = 0; dp < 2; dp++) {
        // in_proj: (TOK x 1024) = hb (TOK x 256) * W (1024 x 256)^T
        int nInW = 1024 * DIMC;
        convert_bf16_kernel<<<(nInW + 255) / 256, 256, 0, stream>>>(
            in_proj_w + (size_t)dp * nInW, wbuf, nInW);
        {
            int tiles = (TOK / 16) * (1024 / 16);
            wmma_gemm_kernel<<<(tiles + 7) / 8, 256, 0, stream>>>(
                hb, wbuf, xz, TOK, 1024, DIMC);
        }
        // depthwise conv1d + SiLU
        conv_silu_kernel<<<(TOK * DI) / 256, 256, 0, stream>>>(
            xz, conv_w + (size_t)dp * DI * 4, conv_b + (size_t)dp * DI, u, ub);
        // x_proj: (TOK x 48) = ub (TOK x 512) * W (48 x 512)^T
        int nXpW = XDBL_W * DI;
        convert_bf16_kernel<<<(nXpW + 255) / 256, 256, 0, stream>>>(
            x_proj_w + (size_t)dp * nXpW, wbuf, nXpW);
        {
            int tiles = (TOK / 16) * (XDBL_W / 16);
            wmma_gemm_kernel<<<(tiles + 7) / 8, 256, 0, stream>>>(
                ub, wbuf, xdbl, TOK, XDBL_W, DI);
        }
        // dt projection + softplus
        dt_kernel<<<(TOK * DI) / 256, 256, 0, stream>>>(
            xdbl, dt_proj_w + (size_t)dp * DI * DTRANK,
            dt_proj_b + (size_t)dp * DI, dtv);
        // selective scan
        scan_kernel<<<(2 * DI) / 256, 256, 0, stream>>>(
            dtv, u, xdbl, xz, A_log + (size_t)dp * DI * DSTATE,
            D_skip + (size_t)dp * DI, y, yb);
        // out_proj: (TOK x 256) = yb (TOK x 512) * W (256 x 512)^T  -> h
        int nOW = DIMC * DI;
        convert_bf16_kernel<<<(nOW + 255) / 256, 256, 0, stream>>>(
            out_proj_w + (size_t)dp * nOW, wbuf, nOW);
        {
            int tiles = (TOK / 16) * (DIMC / 16);
            wmma_gemm_kernel<<<(tiles + 7) / 8, 256, 0, stream>>>(
                yb, wbuf, h, TOK, DIMC, DI);
        }
        // bf16 copy of h for next depth's in_proj
        convert_bf16_kernel<<<(TOK * DIMC + 255) / 256, 256, 0, stream>>>(
            h, hb, TOK * DIMC);
    }

    // Fused conv3d + PReLU + softplus(scale) + residual add -> output
    {
        int tiles = (TOK / 16) * (DIMC / 16);   // 8192 waves
        conv3d_wmma_kernel<<<(tiles + 7) / 8, 256, 0, stream>>>(
            xb, wr, h, prelu_a, scale, out);
    }
}
